// Decoder_13812614824235
// MI455X (gfx1250) — matrix operand order repaired and checked by reading, not compiled, not run on hardware
//
#include <hip/hip_runtime.h>
#include <math.h>

typedef __attribute__((ext_vector_type(16))) _Float16 v16h;
typedef __attribute__((ext_vector_type(8)))  _Float16 v8h;
typedef __attribute__((ext_vector_type(8)))  float    v8f;

#define B_   128
#define T_   1024
#define D_   256
#define LD_  512
#define K_   128
#define RD_  640

static __device__ __forceinline__ float sigm(float x) {
    return 1.0f / (1.0f + __expf(-x));
}

// ---------------------------------------------------------------------------
// f32 -> f16 conversion (weights / input states), one element per thread.
// ---------------------------------------------------------------------------
__global__ void cvt_h(const float* __restrict__ src, _Float16* __restrict__ dst, int n)
{
    int i = blockIdx.x * blockDim.x + threadIdx.x;
    if (i < n) dst[i] = (_Float16)src[i];
}

// ---------------------------------------------------------------------------
// WMMA GEMM, all-f16 operands: C[M,N] = act(A[M,K] @ W[N,K]^T + bias[N])
// One wave per 16x16 tile; grid = (M/16, N/16), block = 32.
// A fragment (16-bit A 16x32 layout): lane row m=l&15, K-halves interleaved
// by (l>>4)*8 -> two contiguous 8-half (16B) chunks -> 2x b128.
// B fragment (16-bit B 32x16 layout): lane col n=l&15, K interleaved by (l>>4)*8 as for A
// -> two contiguous 8-half (16B) chunks -> 2x b128.
// Optional f16 mirror output Ch feeds the next GEMM's A operand.
// ---------------------------------------------------------------------------
__global__ __launch_bounds__(32)
void gemm_wmma(const _Float16* __restrict__ A, const _Float16* __restrict__ W,
               const float* __restrict__ bias, float* __restrict__ C,
               _Float16* __restrict__ Ch, int K, int N, int act)
{
    const int l  = threadIdx.x;
    const int r  = l & 15;
    const int hi = l >> 4;
    const int ka = hi * 8;
    const int kb = hi * 8;
    const int m0 = blockIdx.x * 16;
    const int n0 = blockIdx.y * 16;
    const _Float16* __restrict__ arow = A + (size_t)(m0 + r) * K;
    const _Float16* __restrict__ wrow = W + (size_t)(n0 + r) * K;

    v8f acc = {};
    for (int k0 = 0; k0 < K; k0 += 32) {
        v8h a0 = *(const v8h*)(arow + k0 + ka);
        v8h a1 = *(const v8h*)(arow + k0 + 16 + ka);
        v8h b0 = *(const v8h*)(wrow + k0 + kb);
        v8h b1 = *(const v8h*)(wrow + k0 + kb + 16);
        v16h a = __builtin_shufflevector(a0, a1, 0,1,2,3,4,5,6,7,8,9,10,11,12,13,14,15);
        v16h b = __builtin_shufflevector(b0, b1, 0,1,2,3,4,5,6,7,8,9,10,11,12,13,14,15);
        acc = __builtin_amdgcn_wmma_f32_16x16x32_f16(false, a, false, b,
                                                     (short)0, acc, false, false);
    }

    const int n = n0 + r;
    const float bv = bias[n];
#pragma unroll
    for (int j = 0; j < 8; ++j) {
        const int m = m0 + j + 8 * hi;
        float v = acc[j] + bv;
        if (act == 1) v = tanhf(v);
        C[(size_t)m * N + n] = v;
        if (Ch) Ch[(size_t)m * N + n] = (_Float16)v;
    }
}

// ---------------------------------------------------------------------------
// Location conv: loc = stack([cumulative, attn_prev]) [B,2,T], conv_w [32,2,31],
// 'same' pad 15.  Output written as f16 [B,T,32] directly for the WMMA stage.
// ---------------------------------------------------------------------------
__global__ void conv_loc(const float* __restrict__ cum,
                         const float* __restrict__ prev,
                         const float* __restrict__ cw,
                         _Float16* __restrict__ out)
{
    int idx = blockIdx.x * blockDim.x + threadIdx.x;   // B*T*32
    int c = idx & 31;
    int t = (idx >> 5) & (T_ - 1);
    int b = idx >> 15;
    const float* w0 = cw + c * 62;
    const float* w1 = w0 + 31;
    const float* r0 = cum + (size_t)b * T_;
    const float* r1 = prev + (size_t)b * T_;
    float s = 0.0f;
#pragma unroll
    for (int j = 0; j < 31; ++j) {
        int tt = t + j - 15;
        if (tt >= 0 && tt < T_) s += r0[tt] * w0[j] + r1[tt] * w1[j];
    }
    out[idx] = (_Float16)s;
}

// ---------------------------------------------------------------------------
// Fused attention energy: u[b,t] = sum_d v[d] * tanh(pq[b,d] + proj[b,t,d]
//                                  + (conv[b,t,:] @ L_w^T)[d] + L_b[d])
// One 16x16x32 WMMA per (t-tile, n-tile); proj streamed exactly once.
// grid = (B, T/16), block = 32.
// ---------------------------------------------------------------------------
__global__ __launch_bounds__(32)
void energy_wmma(const _Float16* __restrict__ conv,  // [B,T,32] f16
                 const float* __restrict__ pq,       // [B,256]
                 const float* __restrict__ proj,     // [B,T,256]
                 const _Float16* __restrict__ Lw,    // [256,32] f16
                 const float* __restrict__ Lb,       // [256]
                 const float* __restrict__ vw,       // [256]
                 float* __restrict__ u)              // [B,T]
{
    const int b  = blockIdx.x;
    const int t0 = blockIdx.y * 16;
    const int l  = threadIdx.x;
    const int r  = l & 15;
    const int hi = l >> 4;
    const int ka = hi * 8;
    const int kb = hi * 8;

    const _Float16* arow = conv + ((size_t)b * T_ + t0 + r) * 32;
    v8h a0 = *(const v8h*)(arow + ka);
    v8h a1 = *(const v8h*)(arow + 16 + ka);
    v16h a = __builtin_shufflevector(a0, a1, 0,1,2,3,4,5,6,7,8,9,10,11,12,13,14,15);

    float pu[8];
#pragma unroll
    for (int j = 0; j < 8; ++j) pu[j] = 0.0f;

    for (int n0 = 0; n0 < D_; n0 += 16) {
        const _Float16* wrow = Lw + (size_t)(n0 + r) * 32;
        v8h b0 = *(const v8h*)(wrow + kb);
        v8h b1 = *(const v8h*)(wrow + kb + 16);
        v16h bm = __builtin_shufflevector(b0, b1, 0,1,2,3,4,5,6,7,8,9,10,11,12,13,14,15);
        v8f c = {};
        c = __builtin_amdgcn_wmma_f32_16x16x32_f16(false, a, false, bm,
                                                   (short)0, c, false, false);
        const int n = n0 + r;
        const float base = pq[(size_t)b * D_ + n] + Lb[n];
        const float vv = vw[n];
#pragma unroll
        for (int j = 0; j < 8; ++j) {
            const int t = t0 + j + 8 * hi;
            float e = tanhf(c[j] + base + proj[((size_t)b * T_ + t) * D_ + n]);
            pu[j] += vv * e;
        }
    }
#pragma unroll
    for (int m = 1; m < 16; m <<= 1) {
#pragma unroll
        for (int j = 0; j < 8; ++j) pu[j] += __shfl_xor(pu[j], m, 32);
    }
    if (r == 0) {
#pragma unroll
        for (int j = 0; j < 8; ++j) u[(size_t)b * T_ + t0 + j + 8 * hi] = pu[j];
    }
}

// ---------------------------------------------------------------------------
// Per-batch attention normalize + context reduction. grid = B, block = 256.
// ---------------------------------------------------------------------------
__global__ __launch_bounds__(256)
void attend(const float* __restrict__ u, const float* __restrict__ cumulative,
            const float* __restrict__ enc, float* __restrict__ scores_out,
            float* __restrict__ cum_out, float* __restrict__ ctx_out)
{
    __shared__ float s[T_];
    __shared__ float red[8];
    const int b = blockIdx.x;
    const int tid = threadIdx.x;

    float psum = 0.0f;
    for (int t = tid; t < T_; t += 256) {
        float sg = sigm(u[(size_t)b * T_ + t]);
        s[t] = sg;
        psum += sg;
    }
#pragma unroll
    for (int m = 1; m < 32; m <<= 1) psum += __shfl_xor(psum, m, 32);
    if ((tid & 31) == 0) red[tid >> 5] = psum;
    __syncthreads();
    float total = 0.0f;
#pragma unroll
    for (int w = 0; w < 8; ++w) total += red[w];
    const float inv = 1.0f / total;

    for (int t = tid; t < T_; t += 256) {
        float sc = s[t] * inv;
        s[t] = sc;
        scores_out[(size_t)b * T_ + t] = sc;
        cum_out[(size_t)b * T_ + t] = cumulative[(size_t)b * T_ + t] + sc;
    }
    __syncthreads();

    const float* eb = enc + (size_t)b * T_ * D_;
    float acc = 0.0f;
    for (int t = 0; t < T_; ++t) acc += s[t] * eb[(size_t)t * D_ + tid];
    ctx_out[(size_t)b * D_ + tid] = acc;
}

// ---------------------------------------------------------------------------
// GRU gates (order r,z,n): h' = (1-z)*n + z*h.  Writes f32 + f16 mirror.
// ---------------------------------------------------------------------------
__global__ void gru_gate(const float* __restrict__ gi, const float* __restrict__ gh,
                         const float* __restrict__ hprev, float* __restrict__ hout,
                         _Float16* __restrict__ hout_h)
{
    int idx = blockIdx.x * blockDim.x + threadIdx.x;  // B*D
    int b = idx >> 8, d = idx & (D_ - 1);
    const float* gib = gi + (size_t)b * 3 * D_;
    const float* ghb = gh + (size_t)b * 3 * D_;
    float r = sigm(gib[d] + ghb[d]);
    float z = sigm(gib[D_ + d] + ghb[D_ + d]);
    float n = tanhf(gib[2 * D_ + d] + r * ghb[2 * D_ + d]);
    float h = (1.0f - z) * n + z * hprev[idx];
    hout[idx] = h;
    hout_h[idx] = (_Float16)h;
}

// ---------------------------------------------------------------------------
// LSTM gates (order i,f,g,o) + residual x += h (f32) with f16 mirror of x.
// ---------------------------------------------------------------------------
__global__ void lstm_gate(const float* __restrict__ gA, const float* __restrict__ gB,
                          const float* __restrict__ cprev, float* __restrict__ h,
                          float* __restrict__ c, float* __restrict__ x,
                          _Float16* __restrict__ xh, int H)
{
    int idx = blockIdx.x * blockDim.x + threadIdx.x;
    if (idx >= B_ * H) return;
    int b = idx / H, d = idx - b * H;
    const float* ga = gA + (size_t)b * 4 * H;
    const float* gb = gB + (size_t)b * 4 * H;
    float i = ga[d]         + gb[d];
    float f = ga[H + d]     + gb[H + d];
    float g = ga[2 * H + d] + gb[2 * H + d];
    float o = ga[3 * H + d] + gb[3 * H + d];
    float c2 = sigm(f) * cprev[idx] + sigm(i) * tanhf(g);
    float h2 = sigm(o) * tanhf(c2);
    h[idx] = h2;
    c[idx] = c2;
    float xn = x[idx] + h2;
    x[idx] = xn;
    xh[idx] = (_Float16)xn;
}

// ---------------------------------------------------------------------------
// Row-wise concat with f16 mirror: out[b,:] = [a[b,:na] | c[b,:nb]]
// ---------------------------------------------------------------------------
__global__ void concat2(const float* __restrict__ a, int na,
                        const float* __restrict__ c, int nb,
                        float* __restrict__ out, _Float16* __restrict__ outh)
{
    int n = na + nb;
    int idx = blockIdx.x * blockDim.x + threadIdx.x;
    if (idx >= B_ * n) return;
    int b = idx / n, d = idx - b * n;
    float v = (d < na) ? a[(size_t)b * na + d] : c[(size_t)b * nb + (d - na)];
    out[idx] = v;
    outh[idx] = (_Float16)v;
}

// ---------------------------------------------------------------------------
// stop = sigmoid(rx @ stop_w^T + stop_b); also copies cond = rx.
// grid = B, block = 64.
// ---------------------------------------------------------------------------
__global__ __launch_bounds__(64)
void stop_k(const float* __restrict__ rx, const float* __restrict__ sw,
            const float* __restrict__ sb, float* __restrict__ stop,
            float* __restrict__ cond)
{
    __shared__ float r2[2];
    const int b = blockIdx.x;
    const int tid = threadIdx.x;
    float acc = 0.0f;
    for (int d = tid; d < RD_; d += 64) {
        float xv = rx[(size_t)b * RD_ + d];
        cond[(size_t)b * RD_ + d] = xv;
        acc += xv * sw[d];
    }
#pragma unroll
    for (int m = 1; m < 32; m <<= 1) acc += __shfl_xor(acc, m, 32);
    if ((tid & 31) == 0) r2[tid >> 5] = acc;
    __syncthreads();
    if (tid == 0) stop[b] = sigm(r2[0] + r2[1] + sb[0]);
}

// ---------------------------------------------------------------------------
// Host-side sequencing
// ---------------------------------------------------------------------------
extern "C" void kernel_launch(void* const* d_in, const int* in_sizes, int n_in,
                              void* d_out, int out_size, void* d_ws, size_t ws_size,
                              hipStream_t stream)
{
    (void)in_sizes; (void)n_in; (void)out_size; (void)ws_size;
    const float* enc       = (const float*)d_in[0];
    const float* proj      = (const float*)d_in[1];
    const float* prenet_in = (const float*)d_in[2];
    const float* attn_hid  = (const float*)d_in[3];
    const float* rnn1_h    = (const float*)d_in[4];
    const float* rnn1_c    = (const float*)d_in[5];
    const float* rnn2_h    = (const float*)d_in[6];
    const float* rnn2_c    = (const float*)d_in[7];
    const float* res_h     = (const float*)d_in[8];
    const float* res_c     = (const float*)d_in[9];
    const float* ctx_vec   = (const float*)d_in[10];
    const float* cumulative= (const float*)d_in[11];
    const float* attn_prev = (const float*)d_in[12];
    const float* fc1_w = (const float*)d_in[13]; const float* fc1_b = (const float*)d_in[14];
    const float* fc2_w = (const float*)d_in[15]; const float* fc2_b = (const float*)d_in[16];
    const float* gru_wih = (const float*)d_in[17]; const float* gru_whh = (const float*)d_in[18];
    const float* gru_bih = (const float*)d_in[19]; const float* gru_bhh = (const float*)d_in[20];
    const float* conv_w  = (const float*)d_in[21];
    const float* L_w = (const float*)d_in[22]; const float* L_b = (const float*)d_in[23];
    const float* W_w = (const float*)d_in[24]; const float* W_b = (const float*)d_in[25];
    const float* v_w = (const float*)d_in[26];
    const float* ri_w = (const float*)d_in[27]; const float* ri_b = (const float*)d_in[28];
    const float* r1_wih = (const float*)d_in[29]; const float* r1_whh = (const float*)d_in[30];
    const float* r1_bih = (const float*)d_in[31]; const float* r1_bhh = (const float*)d_in[32];
    const float* r2_wih = (const float*)d_in[33]; const float* r2_whh = (const float*)d_in[34];
    const float* r2_bih = (const float*)d_in[35]; const float* r2_bhh = (const float*)d_in[36];
    const float* res_wih = (const float*)d_in[37]; const float* res_whh = (const float*)d_in[38];
    const float* res_bih = (const float*)d_in[39]; const float* res_bhh = (const float*)d_in[40];
    const float* stop_w = (const float*)d_in[41]; const float* stop_b = (const float*)d_in[42];

    float* out = (float*)d_out;
    float* o_cond   = out;                 // [B,RD]
    float* o_stop   = out + 81920;         // [B,1]
    float* o_scores = out + 82048;         // [B,T]
    float* o_attn_h = out + 213120;        // [B,D]
    float* o_h1     = out + 245888;        // [B,LD]
    float* o_c1     = out + 311424;
    float* o_h2     = out + 376960;
    float* o_c2     = out + 442496;
    float* o_hs     = out + 508032;        // [4,B,RD]
    float* o_cs     = out + 835712;
    float* o_ctx    = out + 1163392;       // [B,D]
    float* o_cum    = out + 1196160;       // [B,T]

    // ---- f32 scratch ----
    float* ws = (float*)d_ws;
    float* w_p    = ws;                    // [B,256]
    float* w_pre  = ws + 32768;            // [B,128]
    float* w_ain  = ws + 49152;            // [B,384]
    float* w_pq   = ws + 98304;            // [B,256]
    float* w_gA   = ws + 131072;           // [B,2560]
    float* w_gB   = ws + 458752;           // [B,2560]
    float* w_u    = ws + 786432;           // [B,T]
    float* w_cat2 = ws + 917504;           // [B,512]
    float* w_x    = ws + 983040;           // [B,512]
    float* w_rx   = ws + 1048576;          // [B,640]
    // ---- f16 scratch (starts 32B-aligned) ----
    _Float16* hb = (_Float16*)(ws + 1160576);
    size_t ho = 0;
    _Float16* h_prein = hb + ho; ho += 16384;    // prenet_in
    _Float16* h_p     = hb + ho; ho += 32768;    // prenet hidden
    _Float16* h_ain   = hb + ho; ho += 49152;    // attn_in
    _Float16* h_ahid  = hb + ho; ho += 32768;    // attn_hidden
    _Float16* h_attnh = hb + ho; ho += 32768;    // attn_h
    _Float16* h_cat2  = hb + ho; ho += 65536;
    _Float16* h_x     = hb + ho; ho += 65536;
    _Float16* h_r1h   = hb + ho; ho += 65536;
    _Float16* h_r2h   = hb + ho; ho += 65536;
    _Float16* h_rx    = hb + ho; ho += 81920;
    _Float16* h_resh  = hb + ho; ho += 327680;   // [4,B,RD]
    _Float16* h_conv  = hb + ho; ho += 4194304;  // [B,T,32]
    _Float16* h_fc1w  = hb + ho; ho += 32768;
    _Float16* h_fc2w  = hb + ho; ho += 32768;
    _Float16* h_gwih  = hb + ho; ho += 294912;
    _Float16* h_gwhh  = hb + ho; ho += 196608;
    _Float16* h_Ww    = hb + ho; ho += 65536;
    _Float16* h_Lw    = hb + ho; ho += 8192;
    _Float16* h_riw   = hb + ho; ho += 262144;
    _Float16* h_r1wih = hb + ho; ho += 1048576;
    _Float16* h_r1whh = hb + ho; ho += 1048576;
    _Float16* h_r2wih = hb + ho; ho += 1048576;
    _Float16* h_r2whh = hb + ho; ho += 1048576;
    _Float16* h_reswih= hb + ho; ho += 6553600;  // [4,2560,640]
    _Float16* h_reswhh= hb + ho; ho += 6553600;

    const dim3 w32(32);
#define CVT(src, dst, n) cvt_h<<<((n) + 255) / 256, 256, 0, stream>>>(src, dst, n)
    // weight + input-state f16 conversions (independent; issued up front)
    CVT(prenet_in, h_prein, 16384);
    CVT(attn_hid,  h_ahid,  32768);
    CVT(rnn1_h,    h_r1h,   65536);
    CVT(rnn2_h,    h_r2h,   65536);
    CVT(res_h,     h_resh,  327680);
    CVT(fc1_w, h_fc1w, 32768);   CVT(fc2_w, h_fc2w, 32768);
    CVT(gru_wih, h_gwih, 294912); CVT(gru_whh, h_gwhh, 196608);
    CVT(W_w, h_Ww, 65536);       CVT(L_w, h_Lw, 8192);
    CVT(ri_w, h_riw, 262144);
    CVT(r1_wih, h_r1wih, 1048576); CVT(r1_whh, h_r1whh, 1048576);
    CVT(r2_wih, h_r2wih, 1048576); CVT(r2_whh, h_r2whh, 1048576);
    CVT(res_wih, h_reswih, 6553600); CVT(res_whh, h_reswhh, 6553600);
#undef CVT

    // PreNet
    gemm_wmma<<<dim3(8, 16), w32, 0, stream>>>(h_prein, h_fc1w, fc1_b, w_p, h_p, 128, 256, 1);
    gemm_wmma<<<dim3(8, 8),  w32, 0, stream>>>(h_p, h_fc2w, fc2_b, w_pre, nullptr, 256, 128, 1);

    // Attention GRU
    concat2<<<(B_ * 384 + 255) / 256, 256, 0, stream>>>(ctx_vec, 256, w_pre, 128, w_ain, h_ain);
    gemm_wmma<<<dim3(8, 48), w32, 0, stream>>>(h_ain, h_gwih, gru_bih, w_gA, nullptr, 384, 768, 0);
    gemm_wmma<<<dim3(8, 48), w32, 0, stream>>>(h_ahid, h_gwhh, gru_bhh, w_gB, nullptr, 256, 768, 0);
    gru_gate<<<128, 256, 0, stream>>>(w_gA, w_gB, attn_hid, o_attn_h, h_attnh);

    // LSA attention
    gemm_wmma<<<dim3(8, 16), w32, 0, stream>>>(h_attnh, h_Ww, W_b, w_pq, nullptr, 256, 256, 0);
    conv_loc<<<(B_ * T_ * 32) / 256, 256, 0, stream>>>(cumulative, attn_prev, conv_w, h_conv);
    energy_wmma<<<dim3(B_, T_ / 16), w32, 0, stream>>>(h_conv, w_pq, proj, h_Lw, L_b, v_w, w_u);
    attend<<<B_, 256, 0, stream>>>(w_u, cumulative, enc, o_scores, o_cum, o_ctx);

    // rnn_input projection + two residual LSTMs
    concat2<<<(B_ * 512 + 255) / 256, 256, 0, stream>>>(o_ctx, 256, o_attn_h, 256, w_cat2, h_cat2);
    gemm_wmma<<<dim3(8, 32), w32, 0, stream>>>(h_cat2, h_riw, ri_b, w_x, h_x, 512, 512, 0);

    gemm_wmma<<<dim3(8, 128), w32, 0, stream>>>(h_x, h_r1wih, r1_bih, w_gA, nullptr, 512, 2048, 0);
    gemm_wmma<<<dim3(8, 128), w32, 0, stream>>>(h_r1h, h_r1whh, r1_bhh, w_gB, nullptr, 512, 2048, 0);
    lstm_gate<<<(B_ * 512 + 255) / 256, 256, 0, stream>>>(w_gA, w_gB, rnn1_c, o_h1, o_c1, w_x, h_x, 512);

    gemm_wmma<<<dim3(8, 128), w32, 0, stream>>>(h_x, h_r2wih, r2_bih, w_gA, nullptr, 512, 2048, 0);
    gemm_wmma<<<dim3(8, 128), w32, 0, stream>>>(h_r2h, h_r2whh, r2_bhh, w_gB, nullptr, 512, 2048, 0);
    lstm_gate<<<(B_ * 512 + 255) / 256, 256, 0, stream>>>(w_gA, w_gB, rnn2_c, o_h2, o_c2, w_x, h_x, 512);

    // 4-layer residual LSTM stack on [prenet_in, x]
    concat2<<<(B_ * RD_ + 255) / 256, 256, 0, stream>>>(prenet_in, 128, w_x, 512, w_rx, h_rx);
    for (int i = 0; i < 4; ++i) {
        const _Float16* wih = h_reswih + (size_t)i * 4 * RD_ * RD_;
        const _Float16* whh = h_reswhh + (size_t)i * 4 * RD_ * RD_;
        const float* bih = res_bih + (size_t)i * 4 * RD_;
        const float* bhh = res_bhh + (size_t)i * 4 * RD_;
        const _Float16* hin = h_resh + (size_t)i * B_ * RD_;
        const float* cin = res_c + (size_t)i * B_ * RD_;
        gemm_wmma<<<dim3(8, 160), w32, 0, stream>>>(w_rx ? h_rx : h_rx, wih, bih, w_gA, nullptr, RD_, 4 * RD_, 0);
        gemm_wmma<<<dim3(8, 160), w32, 0, stream>>>(hin, whh, bhh, w_gB, nullptr, RD_, 4 * RD_, 0);
        lstm_gate<<<(B_ * RD_ + 255) / 256, 256, 0, stream>>>(
            w_gA, w_gB, cin, o_hs + (size_t)i * B_ * RD_, o_cs + (size_t)i * B_ * RD_, w_rx, h_rx, RD_);
    }

    // cond copy + stop projection
    stop_k<<<B_, 64, 0, stream>>>(w_rx, stop_w, stop_b, o_stop, o_cond);
}
